// observation_processing_network_1357209665896
// MI455X (gfx1250) — compile-verified, matrix operand order checked by reading, not run on hardware
//
#include <hip/hip_runtime.h>

typedef _Float16 h16;
typedef __attribute__((ext_vector_type(16))) _Float16 v16h;
typedef __attribute__((ext_vector_type(8)))  _Float16 v8h;
typedef __attribute__((ext_vector_type(8)))  float    v8f;

#define THREADS 256

// Async copy of 16 bytes global -> LDS (gfx1250 async engine, ASYNCcnt-tracked).
static __device__ __forceinline__ void async_b128(unsigned lds_off, const void* g) {
    asm volatile("global_load_async_to_lds_b128 %0, %1, off"
                 :: "v"(lds_off), "v"(g) : "memory");
}
template <int NCNT>
static __device__ __forceinline__ void wait_async_le() {
    if constexpr (NCNT == 0)       asm volatile("s_wait_asynccnt 0x0" ::: "memory");
    else if constexpr (NCNT == 4)  asm volatile("s_wait_asynccnt 0x4" ::: "memory");
    else if constexpr (NCNT == 10) asm volatile("s_wait_asynccnt 0xa" ::: "memory");
}

// ---------------------------------------------------------------------------
// f16 WMMA GEMM: C[M,N] = alpha * A[M,K] @ B^T (+ bias[col]) (relu?)
// A stored [M,K] f16 row-major; B stored [N,K] f16 row-major (pre-transposed).
// REQUIRES: M%16==0, N%(16*NT)==0, K%32==0, rows 16B-aligned.
// One wave per 16x(16*NT) strip. Per K-step the wave async-stages (1+NT)
// 16x32-half tiles into LDS (double-buffered, prefetch pipelined). Fragments
// are two contiguous ds_load_b128 each (fragment K-halves are contiguous).
// ---------------------------------------------------------------------------
template <int NT>
__global__ void wmma_gemm_h(const h16* __restrict__ A, int lda,
                            const h16* __restrict__ B, int ldb,
                            float* __restrict__ C, int ldc,
                            int M, int N, int K,
                            float alpha, const float* __restrict__ bias, int relu) {
    constexpr int TILE_H = 512;                  // halfs per 16x32 tile
    constexpr int BUFH   = (1 + NT) * TILE_H;    // halfs per buffer
    __shared__ h16 smem[2 * BUFH];
    unsigned base = (unsigned)(size_t)(void*)&smem[0];

    int lane = threadIdx.x & 31;
    int hb   = lane >> 4;
    int l    = lane & 15;
    int tm   = blockIdx.y * 16;
    int tn0  = blockIdx.x * (16 * NT);

    auto stage = [&](int buf, int kk) {
        unsigned boff = base + (unsigned)(buf * BUFH) * 2u;
        // A tile: 16 rows x 32 halfs = 64 x 16B chunks, 2 per lane
#pragma unroll
        for (int i = 0; i < 2; ++i) {
            int c = lane + 32 * i;
            int row = c >> 2, col8 = (c & 3) * 8;
            async_b128(boff + (unsigned)c * 16u,
                       A + (size_t)(tm + row) * lda + kk + col8);
        }
        // B tiles (B is [N,K]: tile rows are output columns)
#pragma unroll
        for (int t = 0; t < NT; ++t) {
            int tn = tn0 + 16 * t;
#pragma unroll
            for (int i = 0; i < 2; ++i) {
                int c = lane + 32 * i;
                int row = c >> 2, col8 = (c & 3) * 8;
                async_b128(boff + 1024u + (unsigned)t * 1024u + (unsigned)c * 16u,
                           B + (size_t)(tn + row) * ldb + kk + col8);
            }
        }
    };

    v8f acc[NT] = {};
    stage(0, 0);
    for (int kk = 0; kk < K; kk += 32) {
        int cur = (kk >> 5) & 1;
        bool more = (kk + 32) < K;
        if (more) {
            stage(cur ^ 1, kk + 32);
            wait_async_le<2 + 2 * NT>();         // older batch done (in-order completion)
        } else {
            wait_async_le<0>();
        }

        const h16* As = smem + cur * BUFH;
        const h16* Bs = As + TILE_H;

        // A fragment: two contiguous 16B LDS reads (K halves are contiguous)
        v8h alo = *(const v8h*)(As + l * 32 + hb * 8);
        v8h ahi = *(const v8h*)(As + l * 32 + 16 + hb * 8);
        v16h af = __builtin_shufflevector(alo, ahi, 0, 1, 2, 3, 4, 5, 6, 7,
                                          8, 9, 10, 11, 12, 13, 14, 15);

#pragma unroll
        for (int t = 0; t < NT; ++t) {
            v8h blo = *(const v8h*)(Bs + t * TILE_H + l * 32 + hb * 8);
            v8h bhi = *(const v8h*)(Bs + t * TILE_H + l * 32 + 16 + hb * 8);
            v16h bf = __builtin_shufflevector(blo, bhi, 0, 1, 2, 3, 4, 5, 6, 7,
                                              8, 9, 10, 11, 12, 13, 14, 15);
            acc[t] = __builtin_amdgcn_wmma_f32_16x16x32_f16(false, af, false, bf,
                                                            (short)0, acc[t], false, false);
        }
    }

#pragma unroll
    for (int t = 0; t < NT; ++t) {
        int tn = tn0 + 16 * t;
#pragma unroll
        for (int j = 0; j < 8; ++j) {
            int r = tm + hb * 8 + j;
            float o = alpha * acc[t][j];
            if (bias) o += bias[tn + l];
            if (relu) o = fmaxf(o, 0.f);
            C[(size_t)r * ldc + tn + l] = o;
        }
    }
}

// ---------------------------------------------------------------------------
// Small helpers
// ---------------------------------------------------------------------------
__global__ void fill_kernel(float* __restrict__ p, float v, long count) {
    for (long i = (long)blockIdx.x * blockDim.x + threadIdx.x; i < count;
         i += (long)gridDim.x * blockDim.x)
        p[i] = v;
}

__global__ void copy_kernel(const float* __restrict__ s, float* __restrict__ d, long count) {
    for (long i = (long)blockIdx.x * blockDim.x + threadIdx.x; i < count;
         i += (long)gridDim.x * blockDim.x)
        d[i] = s[i];
}

__global__ void relu_kernel(float* __restrict__ p, long count) {
    for (long i = (long)blockIdx.x * blockDim.x + threadIdx.x; i < count;
         i += (long)gridDim.x * blockDim.x)
        p[i] = fmaxf(p[i], 0.f);
}

__global__ void cvt_h_kernel(const float* __restrict__ src, h16* __restrict__ dst, long count) {
    for (long i = (long)blockIdx.x * blockDim.x + threadIdx.x; i < count;
         i += (long)gridDim.x * blockDim.x)
        dst[i] = (h16)src[i];
}

// src R x C (f32, row-major) -> dst C x R (f16, row-major)   i.e. dst = src^T
__global__ void cvt_trans_h_kernel(const float* __restrict__ src, int R, int C,
                                   h16* __restrict__ dst) {
    long total = (long)R * C;
    for (long idx = (long)blockIdx.x * blockDim.x + threadIdx.x; idx < total;
         idx += (long)gridDim.x * blockDim.x) {
        int r = (int)(idx / C), c = (int)(idx % C);
        dst[(size_t)c * R + r] = (h16)src[idx];
    }
}

__global__ void build_adj_kernel(const int* __restrict__ ei, int Etot, int n,
                                 float* __restrict__ A) {
    int e = blockIdx.x * blockDim.x + threadIdx.x;
    if (e >= Etot) return;
    int s = ei[e], d = ei[Etot + e];
    atomicAdd(&A[(size_t)s * n + d], 1.0f);
}

// out[row] = sum_j Mtx[row,j]*(vec? vec[j]:1) + (biasScalar? *biasScalar:0)
__global__ void row_reduce_kernel(const float* __restrict__ Mtx,
                                  const float* __restrict__ vec,
                                  const float* __restrict__ biasScalar,
                                  float* __restrict__ out, int n) {
    __shared__ float sd[THREADS];
    int row = blockIdx.x;
    float s = 0.f;
    for (int j = threadIdx.x; j < n; j += blockDim.x)
        s += Mtx[(size_t)row * n + j] * (vec ? vec[j] : 1.0f);
    sd[threadIdx.x] = s;
    __syncthreads();
    for (int st = THREADS / 2; st > 0; st >>= 1) {
        if (threadIdx.x < st) sd[threadIdx.x] += sd[threadIdx.x + st];
        __syncthreads();
    }
    if (threadIdx.x == 0) out[row] = sd[0] + (biasScalar ? biasScalar[0] : 0.f);
}

__global__ void dinv_kernel(const float* __restrict__ deg, float* __restrict__ dinv,
                            float* __restrict__ d2, int n) {
    int i = blockIdx.x * blockDim.x + threadIdx.x;
    if (i >= n) return;
    float dg = deg[i];
    dinv[i] = dg > 0.f ? rsqrtf(fmaxf(dg, 1e-12f)) : 0.f;
    d2[i]   = rsqrtf(dg + 1.0f);
}

__global__ void ahat_kernel(const float* __restrict__ A, const float* __restrict__ d2,
                            float* __restrict__ Ahat, h16* __restrict__ Ahath, int n) {
    long nn = (long)n * n;
    for (long idx = (long)blockIdx.x * blockDim.x + threadIdx.x; idx < nn;
         idx += (long)gridDim.x * blockDim.x) {
        int i = (int)(idx / n), j = (int)(idx % n);
        float a = A[idx] + (i == j ? 1.0f : 0.0f);
        float v = d2[i] * a * d2[j];
        Ahat[idx]  = v;
        Ahath[idx] = (h16)v;
    }
}

// Vout[:,c] = Vin[:,c] + dinv .* (A @ (dinv .* Vin[:,c]))    (M = 2I - L applied)
__global__ void pow_mv_kernel(const float* __restrict__ A, const float* __restrict__ dinv,
                              const float* __restrict__ Vin, float* __restrict__ Vout, int n) {
    __shared__ float s0[THREADS];
    __shared__ float s1[THREADS];
    int row = blockIdx.x;
    float a0 = 0.f, a1 = 0.f;
    for (int j = threadIdx.x; j < n; j += blockDim.x) {
        float w = A[(size_t)row * n + j] * dinv[j];
        a0 += w * Vin[j];
        a1 += w * Vin[n + j];
    }
    s0[threadIdx.x] = a0; s1[threadIdx.x] = a1;
    __syncthreads();
    for (int st = THREADS / 2; st > 0; st >>= 1) {
        if (threadIdx.x < st) { s0[threadIdx.x] += s0[threadIdx.x + st];
                                s1[threadIdx.x] += s1[threadIdx.x + st]; }
        __syncthreads();
    }
    if (threadIdx.x == 0) {
        Vout[row]     = Vin[row]     + dinv[row] * s0[0];
        Vout[n + row] = Vin[n + row] + dinv[row] * s1[0];
    }
}

__global__ void gs_kernel(float* __restrict__ V, int n) {
    __shared__ float sd[1024];
    __shared__ float bcast;
    int t = threadIdx.x, nt = blockDim.x;
    float s = 0.f;
    for (int i = t; i < n; i += nt) s += V[i] * V[i];
    sd[t] = s; __syncthreads();
    for (int st = nt / 2; st > 0; st >>= 1) { if (t < st) sd[t] += sd[t + st]; __syncthreads(); }
    if (t == 0) bcast = rsqrtf(sd[0] + 1e-30f);
    __syncthreads();
    float inv0 = bcast;
    for (int i = t; i < n; i += nt) V[i] *= inv0;
    __syncthreads();
    s = 0.f;
    for (int i = t; i < n; i += nt) s += V[i] * V[n + i];
    sd[t] = s; __syncthreads();
    for (int st = nt / 2; st > 0; st >>= 1) { if (t < st) sd[t] += sd[t + st]; __syncthreads(); }
    if (t == 0) bcast = sd[0];
    __syncthreads();
    float proj = bcast;
    for (int i = t; i < n; i += nt) V[n + i] -= proj * V[i];
    __syncthreads();
    s = 0.f;
    for (int i = t; i < n; i += nt) s += V[n + i] * V[n + i];
    sd[t] = s; __syncthreads();
    for (int st = nt / 2; st > 0; st >>= 1) { if (t < st) sd[t] += sd[t + st]; __syncthreads(); }
    if (t == 0) bcast = rsqrtf(sd[0] + 1e-30f);
    __syncthreads();
    float inv1 = bcast;
    for (int i = t; i < n; i += nt) V[n + i] *= inv1;
}

__global__ void init_eig_kernel(float* __restrict__ V, int n) {
    int i = blockIdx.x * blockDim.x + threadIdx.x;
    if (i >= n) return;
    V[i] = 1.0f;
    unsigned h = (unsigned)i * 2654435761u;
    h ^= h >> 13; h *= 0x5bd1e995u; h ^= h >> 15;
    V[n + i] = ((float)(h & 0xFFFFu) / 65535.0f) * 2.0f - 1.0f;
}

__global__ void xc_kernel(const float* __restrict__ x, const float* __restrict__ V,
                          float* __restrict__ xc, int n) {
    int i = blockIdx.x * blockDim.x + threadIdx.x;
    if (i >= n) return;
    xc[i * 5 + 0] = x[i * 3 + 0];
    xc[i * 5 + 1] = x[i * 3 + 1];
    xc[i * 5 + 2] = x[i * 3 + 2];
    xc[i * 5 + 3] = V[i];
    xc[i * 5 + 4] = V[n + i];
}

// Y[i*ldy+o] = X[i*ldx + 0..kin) @ Wm[kin,kout] + bias (optional) (relu optional)
__global__ void small_gemm_kernel(const float* __restrict__ X, int m, int kin, int ldx,
                                  const float* __restrict__ Wm, int kout,
                                  const float* __restrict__ bias,
                                  float* __restrict__ Y, int ldy, int relu) {
    long idx = (long)blockIdx.x * blockDim.x + threadIdx.x;
    if (idx >= (long)m * kout) return;
    int i = (int)(idx / kout), o = (int)(idx % kout);
    float s = bias ? bias[o] : 0.f;
    for (int k2 = 0; k2 < kin; ++k2)
        s += X[(size_t)i * ldx + k2] * Wm[(size_t)k2 * kout + o];
    if (relu) s = fmaxf(s, 0.f);
    Y[(size_t)i * ldy + o] = s;
}

// Same, but writes f16 (for WMMA operand production)
__global__ void small_gemm_h_kernel(const float* __restrict__ X, int m, int kin, int ldx,
                                    const float* __restrict__ Wm, int kout,
                                    const float* __restrict__ bias,
                                    h16* __restrict__ Y, int ldy, int relu) {
    long idx = (long)blockIdx.x * blockDim.x + threadIdx.x;
    if (idx >= (long)m * kout) return;
    int i = (int)(idx / kout), o = (int)(idx % kout);
    float s = bias ? bias[o] : 0.f;
    for (int k2 = 0; k2 < kin; ++k2)
        s += X[(size_t)i * ldx + k2] * Wm[(size_t)k2 * kout + o];
    if (relu) s = fmaxf(s, 0.f);
    Y[(size_t)i * ldy + o] = (h16)s;
}

static __device__ __forceinline__ void edge_sd(const int* ei, int Etot, int e, int* s, int* d) {
    if (e < Etot) { *s = ei[e]; *d = ei[Etot + e]; }
    else          { *s = e - Etot; *d = e - Etot; }        // self loops
}

static __device__ __forceinline__ void atomicMaxF(float* addr, float val) {
    unsigned int* ua = (unsigned int*)addr;
    unsigned int old = *ua;
    while (true) {
        float f = __uint_as_float(old);
        if (f >= val) break;
        unsigned int assumed = old;
        old = atomicCAS(ua, assumed, __float_as_uint(val));
        if (old == assumed) break;
    }
}

__global__ void gat_score_kernel(const int* __restrict__ ei, int Etot, int E2,
                                 const float* __restrict__ al, const float* __restrict__ ad,
                                 float* __restrict__ esc) {
    int e = blockIdx.x * blockDim.x + threadIdx.x;
    if (e >= E2) return;
    int s, d; edge_sd(ei, Etot, e, &s, &d);
    float v = al[s] + ad[d];
    esc[e] = (v > 0.f) ? v : 0.2f * v;                     // leaky_relu 0.2
}

__global__ void dot_score_kernel(const int* __restrict__ ei, int Etot, int E2,
                                 const float* __restrict__ tq, const float* __restrict__ tk,
                                 int F, float scale, float* __restrict__ esc) {
    int e = blockIdx.x * blockDim.x + threadIdx.x;
    if (e >= E2) return;
    int s, d; edge_sd(ei, Etot, e, &s, &d);
    float v = 0.f;
    for (int f = 0; f < F; ++f) v += tq[d * F + f] * tk[s * F + f];
    esc[e] = v * scale;
}

__global__ void seg_max_kernel(const int* __restrict__ ei, int Etot, int E2,
                               const float* __restrict__ esc, float* __restrict__ segmax) {
    int e = blockIdx.x * blockDim.x + threadIdx.x;
    if (e >= E2) return;
    int s, d; edge_sd(ei, Etot, e, &s, &d);
    atomicMaxF(&segmax[d], esc[e]);
}

__global__ void edge_exp_kernel(const int* __restrict__ ei, int Etot, int E2,
                                float* __restrict__ esc, const float* __restrict__ segmax,
                                float* __restrict__ segsum) {
    int e = blockIdx.x * blockDim.x + threadIdx.x;
    if (e >= E2) return;
    int s, d; edge_sd(ei, Etot, e, &s, &d);
    float w = expf(esc[e] - segmax[d]);
    esc[e] = w;
    atomicAdd(&segsum[d], w);
}

__global__ void init_bias_kernel(float* __restrict__ out, const float* __restrict__ bias,
                                 int n, int F) {
    int idx = blockIdx.x * blockDim.x + threadIdx.x;
    if (idx >= n * F) return;
    out[idx] = bias[idx % F];
}

__global__ void edge_accum_kernel(const int* __restrict__ ei, int Etot, int E2,
                                  const float* __restrict__ esc, const float* __restrict__ segsum,
                                  const float* __restrict__ hw, int F, float* __restrict__ out) {
    int e = blockIdx.x * blockDim.x + threadIdx.x;
    if (e >= E2) return;
    int s, d; edge_sd(ei, Etot, e, &s, &d);
    float a = esc[e] / (segsum[d] + 1e-16f);
    for (int f = 0; f < F; ++f) atomicAdd(&out[d * F + f], a * hw[s * F + f]);
}

// row softmax on f32 S, also writes f16 copy Sh (WMMA A operand for attn@v)
__global__ void softmax_rows_kernel(float* __restrict__ S, h16* __restrict__ Sh, int n) {
    __shared__ float sd[THREADS];
    int row = blockIdx.x;
    float m = -3.0e38f;
    for (int j = threadIdx.x; j < n; j += blockDim.x) m = fmaxf(m, S[(size_t)row * n + j]);
    sd[threadIdx.x] = m; __syncthreads();
    for (int st = THREADS / 2; st > 0; st >>= 1) {
        if (threadIdx.x < st) sd[threadIdx.x] = fmaxf(sd[threadIdx.x], sd[threadIdx.x + st]);
        __syncthreads();
    }
    m = sd[0]; __syncthreads();
    float s = 0.f;
    for (int j = threadIdx.x; j < n; j += blockDim.x) {
        float w = expf(S[(size_t)row * n + j] - m);
        S[(size_t)row * n + j] = w;
        s += w;
    }
    sd[threadIdx.x] = s; __syncthreads();
    for (int st = THREADS / 2; st > 0; st >>= 1) {
        if (threadIdx.x < st) sd[threadIdx.x] += sd[threadIdx.x + st];
        __syncthreads();
    }
    float inv = 1.0f / sd[0];
    for (int j = threadIdx.x; j < n; j += blockDim.x) {
        float w = S[(size_t)row * n + j] * inv;
        S[(size_t)row * n + j]  = w;
        Sh[(size_t)row * n + j] = (h16)w;
    }
}

__global__ void argmax_kernel(const float* __restrict__ u, int n, float* __restrict__ out2) {
    __shared__ float sv[THREADS];
    __shared__ int   si[THREADS];
    float m = -3.0e38f; int mi = 0;
    for (int j = threadIdx.x; j < n; j += blockDim.x) {
        float v = u[j];
        if (v > m) { m = v; mi = j; }
    }
    sv[threadIdx.x] = m; si[threadIdx.x] = mi; __syncthreads();
    for (int st = THREADS / 2; st > 0; st >>= 1) {
        if (threadIdx.x < st && sv[threadIdx.x + st] > sv[threadIdx.x]) {
            sv[threadIdx.x] = sv[threadIdx.x + st];
            si[threadIdx.x] = si[threadIdx.x + st];
        }
        __syncthreads();
    }
    if (threadIdx.x == 0) { out2[0] = sv[0]; ((int*)out2)[1] = si[0]; }
}

__global__ void finalize_kernel(float* __restrict__ logits, const float* __restrict__ mask,
                                const float* __restrict__ out2, float* __restrict__ dout, int n) {
    float umax = out2[0];
    int crit = ((const int*)out2)[1];
    float add = ((0.8f - umax) < 0.2f) ? 10.0f : 0.0f;
    long nn = (long)n * n;
    for (long idx = (long)blockIdx.x * blockDim.x + threadIdx.x; idx < nn;
         idx += (long)gridDim.x * blockDim.x) {
        int i = (int)(idx / n), j = (int)(idx % n);
        float l = logits[idx];
        if (i == crit) l += add;
        logits[idx] = l;
        dout[idx] = l * mask[j];
    }
}

__global__ void set_scalar_kernel(float* __restrict__ p, const float* __restrict__ src) {
    if (threadIdx.x == 0 && blockIdx.x == 0) p[0] = src[0];
}

__global__ void value_rows_kernel(const float* __restrict__ logits, const float* __restrict__ Wc,
                                  float* __restrict__ vslot, int n) {
    __shared__ float sd[THREADS];
    int row = blockIdx.x;
    float s = 0.f;
    for (int j = threadIdx.x; j < n; j += blockDim.x)
        s += logits[(size_t)row * n + j] * Wc[j];
    sd[threadIdx.x] = s; __syncthreads();
    for (int st = THREADS / 2; st > 0; st >>= 1) {
        if (threadIdx.x < st) sd[threadIdx.x] += sd[threadIdx.x + st];
        __syncthreads();
    }
    if (threadIdx.x == 0) atomicAdd(vslot, sd[0] / (float)n);
}

// ---------------------------------------------------------------------------
extern "C" void kernel_launch(void* const* d_in, const int* in_sizes, int n_in,
                              void* d_out, int out_size, void* d_ws, size_t ws_size,
                              hipStream_t stream) {
    const float* x     = (const float*)d_in[0];
    const int*   ei    = (const int*)d_in[1];
    const float* mask  = (const float*)d_in[2];
    const float* Wg1   = (const float*)d_in[3];
    const float* bg1   = (const float*)d_in[4];
    const float* Wg2   = (const float*)d_in[5];
    const float* bg2   = (const float*)d_in[6];
    const float* Wa1   = (const float*)d_in[7];
    const float* asrc1 = (const float*)d_in[8];
    const float* adst1 = (const float*)d_in[9];
    const float* ba1   = (const float*)d_in[10];
    const float* Wa2   = (const float*)d_in[11];
    const float* asrc2 = (const float*)d_in[12];
    const float* adst2 = (const float*)d_in[13];
    const float* ba2   = (const float*)d_in[14];
    const float* Wq    = (const float*)d_in[15];
    const float* bq    = (const float*)d_in[16];
    const float* Wk    = (const float*)d_in[17];
    const float* bk    = (const float*)d_in[18];
    const float* Wv    = (const float*)d_in[19];
    const float* bv    = (const float*)d_in[20];
    const float* Wo    = (const float*)d_in[21];
    const float* bo    = (const float*)d_in[22];
    const float* Wtq   = (const float*)d_in[23];
    const float* btq   = (const float*)d_in[24];
    const float* Wtk   = (const float*)d_in[25];
    const float* btk   = (const float*)d_in[26];
    const float* Wtv   = (const float*)d_in[27];
    const float* btv   = (const float*)d_in[28];
    const float* Wts   = (const float*)d_in[29];
    const float* bts   = (const float*)d_in[30];
    const float* W1    = (const float*)d_in[31];
    const float* b1    = (const float*)d_in[32];
    const float* W2    = (const float*)d_in[33];
    const float* b2    = (const float*)d_in[34];
    const float* Wc    = (const float*)d_in[35];
    const float* bc    = (const float*)d_in[36];
    (void)n_in; (void)out_size; (void)ws_size;

    const int n    = in_sizes[0] / 3;        // 2048
    const int Etot = in_sizes[1] / 2;        // 131072 (symmetrized)
    const int E2   = Etot + n;               // + self loops
    const size_t nn = (size_t)n * n;
    float* dout = (float*)d_out;

    float* Wf = (float*)d_ws;
    size_t off = 0;
    auto alloc = [&](size_t c) { float* p = Wf + off; off += c; return p; };
    float* A      = alloc(nn);
    float* Ahat   = alloc(nn);
    float* S      = alloc(nn);
    float* logits = alloc(nn);
    h16*   Sh     = (h16*)alloc(nn / 2);     // f16 softmax(S)  (A operand)
    h16*   Ahath  = (h16*)alloc(nn / 2);     // f16 Ahat        (A operand)
    float* deg    = alloc(n);
    float* dinv   = alloc(n);
    float* d2     = alloc(n);
    float* V      = alloc(2 * (size_t)n);
    float* Vt     = alloc(2 * (size_t)n);
    float* xc     = alloc(5 * (size_t)n);
    float* hwA    = alloc(8 * (size_t)n);
    float* al     = alloc(n);
    float* ad     = alloc(n);
    float* h1     = alloc(8 * (size_t)n);
    float* hB     = alloc(5 * (size_t)n);
    h16*   qh     = (h16*)alloc(16 * (size_t)n);  // q padded [n,32] f16
    h16*   kh     = (h16*)alloc(16 * (size_t)n);  // k padded [n,32] f16 ([N,K])
    float* vb5    = alloc(5 * (size_t)n);         // v [n,5] f32
    h16*   vT     = (h16*)alloc(8 * (size_t)n);   // v^T padded [16,n] f16 ([N,K])
    float* av     = alloc(16 * (size_t)n);        // attn@v padded [n,16]
    float* h2     = alloc(5 * (size_t)n);
    float* tq     = alloc(5 * (size_t)n);
    float* tk     = alloc(5 * (size_t)n);
    float* tv     = alloc(5 * (size_t)n);
    float* xt     = alloc(5 * (size_t)n);
    h16*   hmh    = (h16*)alloc(32 * (size_t)n);  // hm [n,64] f16 (A operand)
    h16*   W2t    = (h16*)alloc(32 * (size_t)n);  // W2^T [n,64] f16 ([N,K])
    float* hw1    = alloc(16 * (size_t)n);
    h16*   hw1T   = (h16*)alloc(8 * (size_t)n);   // hw1^T [16,n] f16 ([N,K])
    float* z      = alloc(16 * (size_t)n);
    float* w2col  = alloc(n);
    float* u      = alloc(n);
    float* esc    = alloc(E2);
    float* segmax = alloc(n);
    float* segsum = alloc(n);
    float* out2   = alloc(2);

    auto cdiv = [](long a, long b) { return (unsigned)((a + b - 1) / b); };
    const float inv_sqrt5 = 0.44721359549995794f;

    // ---- adjacency, degrees, Ahat (f32 + f16) -----------------------------
    fill_kernel<<<4096, THREADS, 0, stream>>>(A, 0.f, (long)nn);
    build_adj_kernel<<<cdiv(Etot, THREADS), THREADS, 0, stream>>>(ei, Etot, n, A);
    row_reduce_kernel<<<n, THREADS, 0, stream>>>(A, nullptr, nullptr, deg, n);
    dinv_kernel<<<cdiv(n, THREADS), THREADS, 0, stream>>>(deg, dinv, d2, n);
    ahat_kernel<<<4096, THREADS, 0, stream>>>(A, d2, Ahat, Ahath, n);

    // ---- Laplacian eigenvectors via orthogonal iteration on M = 2I - L ----
    init_eig_kernel<<<cdiv(n, THREADS), THREADS, 0, stream>>>(V, n);
    float* Va = V; float* Vb = Vt;
    for (int it = 0; it < 30; ++it) {
        pow_mv_kernel<<<n, THREADS, 0, stream>>>(A, dinv, Va, Vb, n);
        gs_kernel<<<1, 1024, 0, stream>>>(Vb, n);
        float* tmp = Va; Va = Vb; Vb = tmp;
    }
    xc_kernel<<<cdiv(n, THREADS), THREADS, 0, stream>>>(x, Va, xc, n);

    // ---- GAT layer 1 (F=8) ------------------------------------------------
    small_gemm_kernel<<<cdiv((long)n * 8, THREADS), THREADS, 0, stream>>>(xc, n, 5, 5, Wa1, 8, nullptr, hwA, 8, 0);
    small_gemm_kernel<<<cdiv(n, THREADS), THREADS, 0, stream>>>(hwA, n, 8, 8, asrc1, 1, nullptr, al, 1, 0);
    small_gemm_kernel<<<cdiv(n, THREADS), THREADS, 0, stream>>>(hwA, n, 8, 8, adst1, 1, nullptr, ad, 1, 0);
    gat_score_kernel<<<cdiv(E2, THREADS), THREADS, 0, stream>>>(ei, Etot, E2, al, ad, esc);
    fill_kernel<<<cdiv(n, THREADS), THREADS, 0, stream>>>(segmax, -3.0e38f, n);
    fill_kernel<<<cdiv(n, THREADS), THREADS, 0, stream>>>(segsum, 0.f, n);
    seg_max_kernel<<<cdiv(E2, THREADS), THREADS, 0, stream>>>(ei, Etot, E2, esc, segmax);
    edge_exp_kernel<<<cdiv(E2, THREADS), THREADS, 0, stream>>>(ei, Etot, E2, esc, segmax, segsum);
    init_bias_kernel<<<cdiv((long)n * 8, THREADS), THREADS, 0, stream>>>(h1, ba1, n, 8);
    edge_accum_kernel<<<cdiv(E2, THREADS), THREADS, 0, stream>>>(ei, Etot, E2, esc, segsum, hwA, 8, h1);
    relu_kernel<<<cdiv((long)n * 8, THREADS), THREADS, 0, stream>>>(h1, (long)n * 8);

    // ---- GAT layer 2 (F=5) ------------------------------------------------
    small_gemm_kernel<<<cdiv((long)n * 5, THREADS), THREADS, 0, stream>>>(h1, n, 8, 8, Wa2, 5, nullptr, hwA, 5, 0);
    small_gemm_kernel<<<cdiv(n, THREADS), THREADS, 0, stream>>>(hwA, n, 5, 5, asrc2, 1, nullptr, al, 1, 0);
    small_gemm_kernel<<<cdiv(n, THREADS), THREADS, 0, stream>>>(hwA, n, 5, 5, adst2, 1, nullptr, ad, 1, 0);
    gat_score_kernel<<<cdiv(E2, THREADS), THREADS, 0, stream>>>(ei, Etot, E2, al, ad, esc);
    fill_kernel<<<cdiv(n, THREADS), THREADS, 0, stream>>>(segmax, -3.0e38f, n);
    fill_kernel<<<cdiv(n, THREADS), THREADS, 0, stream>>>(segsum, 0.f, n);
    seg_max_kernel<<<cdiv(E2, THREADS), THREADS, 0, stream>>>(ei, Etot, E2, esc, segmax);
    edge_exp_kernel<<<cdiv(E2, THREADS), THREADS, 0, stream>>>(ei, Etot, E2, esc, segmax, segsum);
    init_bias_kernel<<<cdiv((long)n * 5, THREADS), THREADS, 0, stream>>>(hB, ba2, n, 5);
    edge_accum_kernel<<<cdiv(E2, THREADS), THREADS, 0, stream>>>(ei, Etot, E2, esc, segsum, hwA, 5, hB);

    // ---- Multi-head attention (1 head, d=5) over all nodes -----------------
    fill_kernel<<<cdiv((long)n * 16, THREADS), THREADS, 0, stream>>>((float*)qh, 0.f, (long)n * 16);
    fill_kernel<<<cdiv((long)n * 16, THREADS), THREADS, 0, stream>>>((float*)kh, 0.f, (long)n * 16);
    fill_kernel<<<cdiv((long)n * 8, THREADS), THREADS, 0, stream>>>((float*)vT, 0.f, (long)n * 8);
    small_gemm_h_kernel<<<cdiv((long)n * 5, THREADS), THREADS, 0, stream>>>(hB, n, 5, 5, Wq, 5, bq, qh, 32, 0);
    small_gemm_h_kernel<<<cdiv((long)n * 5, THREADS), THREADS, 0, stream>>>(hB, n, 5, 5, Wk, 5, bk, kh, 32, 0);
    small_gemm_kernel<<<cdiv((long)n * 5, THREADS), THREADS, 0, stream>>>(hB, n, 5, 5, Wv, 5, bv, vb5, 5, 0);
    cvt_trans_h_kernel<<<cdiv((long)n * 5, THREADS), THREADS, 0, stream>>>(vb5, n, 5, vT);
    {
        dim3 g(cdiv(n, 64), cdiv(n, 16));
        wmma_gemm_h<4><<<g, 32, 0, stream>>>(qh, 32, kh, 32, S, n,
                                             n, n, 32, inv_sqrt5, nullptr, 0);
    }
    softmax_rows_kernel<<<n, THREADS, 0, stream>>>(S, Sh, n);
    {
        dim3 g(1, cdiv(n, 16));
        wmma_gemm_h<1><<<g, 32, 0, stream>>>(Sh, n, vT, n, av, 16,
                                             n, 16, n, 1.0f, nullptr, 0);
    }
    small_gemm_kernel<<<cdiv((long)n * 5, THREADS), THREADS, 0, stream>>>(av, n, 5, 16, Wo, 5, bo, h2, 5, 0);

    // ---- TransformerConv ---------------------------------------------------
    small_gemm_kernel<<<cdiv((long)n * 5, THREADS), THREADS, 0, stream>>>(h2, n, 5, 5, Wtq, 5, btq, tq, 5, 0);
    small_gemm_kernel<<<cdiv((long)n * 5, THREADS), THREADS, 0, stream>>>(h2, n, 5, 5, Wtk, 5, btk, tk, 5, 0);
    small_gemm_kernel<<<cdiv((long)n * 5, THREADS), THREADS, 0, stream>>>(h2, n, 5, 5, Wtv, 5, btv, tv, 5, 0);
    small_gemm_kernel<<<cdiv((long)n * 5, THREADS), THREADS, 0, stream>>>(h2, n, 5, 5, Wts, 5, bts, xt, 5, 0); // skip term
    dot_score_kernel<<<cdiv(E2, THREADS), THREADS, 0, stream>>>(ei, Etot, E2, tq, tk, 5, inv_sqrt5, esc);
    fill_kernel<<<cdiv(n, THREADS), THREADS, 0, stream>>>(segmax, -3.0e38f, n);
    fill_kernel<<<cdiv(n, THREADS), THREADS, 0, stream>>>(segsum, 0.f, n);
    seg_max_kernel<<<cdiv(E2, THREADS), THREADS, 0, stream>>>(ei, Etot, E2, esc, segmax);
    edge_exp_kernel<<<cdiv(E2, THREADS), THREADS, 0, stream>>>(ei, Etot, E2, esc, segmax, segsum);
    edge_accum_kernel<<<cdiv(E2, THREADS), THREADS, 0, stream>>>(ei, Etot, E2, esc, segsum, tv, 5, xt);

    // ---- MLP head: hm = relu(xt @ W1[:5] + b1); logits = hm @ W2 + b2 ------
    small_gemm_h_kernel<<<cdiv((long)n * 64, THREADS), THREADS, 0, stream>>>(xt, n, 5, 5, W1, 64, b1, hmh, 64, 1);
    cvt_trans_h_kernel<<<cdiv((long)n * 64, THREADS), THREADS, 0, stream>>>(W2, 64, n, W2t);
    {
        dim3 g(cdiv(n, 64), cdiv(n, 16));
        wmma_gemm_h<4><<<g, 32, 0, stream>>>(hmh, 64, W2t, 64, logits, n,
                                             n, n, 64, 1.0f, b2, 0);
    }

    // ---- GCN uncertainty head: u = gcn(xt) --------------------------------
    small_gemm_kernel<<<cdiv((long)n * 16, THREADS), THREADS, 0, stream>>>(xt, n, 5, 5, Wg1, 16, nullptr, hw1, 16, 0);
    cvt_trans_h_kernel<<<cdiv((long)n * 16, THREADS), THREADS, 0, stream>>>(hw1, n, 16, hw1T);
    {
        dim3 g(1, cdiv(n, 16));
        wmma_gemm_h<1><<<g, 32, 0, stream>>>(Ahath, n, hw1T, n, z, 16,
                                             n, 16, n, 1.0f, bg1, 1);
    }
    small_gemm_kernel<<<cdiv(n, THREADS), THREADS, 0, stream>>>(z, n, 16, 16, Wg2, 1, nullptr, w2col, 1, 0);
    row_reduce_kernel<<<n, THREADS, 0, stream>>>(Ahat, w2col, bg2, u, n);

    // ---- safe-logit adjustment, masking, value ----------------------------
    argmax_kernel<<<1, THREADS, 0, stream>>>(u, n, out2);
    finalize_kernel<<<4096, THREADS, 0, stream>>>(logits, mask, out2, dout, n);
    set_scalar_kernel<<<1, 32, 0, stream>>>(dout + nn, bc);
    value_rows_kernel<<<n, THREADS, 0, stream>>>(logits, Wc, dout + nn, n);
    copy_kernel<<<cdiv((long)n * 5, THREADS), THREADS, 0, stream>>>(xc, dout + nn + 1, (long)n * 5);
}
